// ZetaAttention_26757646254404
// MI455X (gfx1250) — matrix operand order repaired and checked by reading, not compiled, not run on hardware
//
#include <hip/hip_runtime.h>
#include <hip/hip_bf16.h>

// MI455X / gfx1250, wave32. bf16 WMMA pipeline for ZetaAttention.
// Workspace layout (needs 64 MB):
//   [0,8MB)    xb   : x as bf16            (B*S x D)
//   [8,10)    wqb, [10,12) wkb, [12,14) wvb, [14,16) wob : weights bf16
//   [16,24)   qb   : Q bf16 (b,h,s,dk)
//   [24,32)   kb   : K bf16 (b,h,s,dk)
//   [32,40)   vtb  : V bf16 TRANSPOSED (b,h,dk,s)  -> contiguous B-frag loads
//   [40,48)   ab   : attention output bf16 (b,s,d)
//   [48,64)   Mk   : memory kernel f32 (S x S), computed once, reused B*H times

typedef __attribute__((ext_vector_type(16))) __bf16 v16bf;
typedef __attribute__((ext_vector_type(8)))  __bf16 v8bf;
typedef __attribute__((ext_vector_type(8)))  float  v8f;
typedef __attribute__((ext_vector_type(4)))  float  v4f;
typedef __attribute__((ext_vector_type(4)))  int    v4i_;

#define S_  2048
#define D_  1024
#define H_  16
#define DK_ 64
#define B_  2
#define V_  4096
#define NZ_ 10

__device__ __forceinline__ v16bf join8(v8bf lo, v8bf hi) {
  v16bf r;
#pragma unroll
  for (int i = 0; i < 8; ++i) { r[i] = lo[i]; r[i + 8] = hi[i]; }
  return r;
}

// ---- gfx1250 async global->LDS copy (16B), ASYNCcnt-tracked; sync fallback.
__device__ __forceinline__ void async_copy16(const __bf16* g, __bf16* l) {
#if __has_builtin(__builtin_amdgcn_global_load_async_to_lds_b128)
  __builtin_amdgcn_global_load_async_to_lds_b128(
      (__attribute__((address_space(1))) v4i_*)g,
      (__attribute__((address_space(3))) v4i_*)l, 0, 0);
#else
  *(v8bf*)l = *(const v8bf*)g;
#endif
}
__device__ __forceinline__ void async_wait_all() {
#if __has_builtin(__builtin_amdgcn_global_load_async_to_lds_b128)
  asm volatile("s_wait_asynccnt 0x0" ::: "memory");
#endif
}

// ---------------- f32 -> bf16 conversion (4 elems/thread) ----------------
__global__ void cvt_f32_bf16(const float* __restrict__ src,
                             __bf16* __restrict__ dst, int n4) {
  int i = blockIdx.x * 256 + threadIdx.x;
  if (i >= n4) return;
  v4f v = *(const v4f*)&src[(size_t)i * 4];
  __bf16 o[4];
#pragma unroll
  for (int e = 0; e < 4; ++e) o[e] = (__bf16)v[e];
  *(unsigned long long*)&dst[(size_t)i * 4] = *(unsigned long long*)o;
}

// ---------------- memory kernel M (S x S) ----------------
__global__ void build_memkernel(const float* __restrict__ gamma,
                                float* __restrict__ M) {
  int idx = blockIdx.x * 256 + threadIdx.x;
  if (idx >= S_ * S_) return;
  int i = idx >> 11, j = idx & (S_ - 1);
  float delta = fabsf((float)(i - j));
  float decay = __expf(-delta);            // KAPPA = 1.0
  float ld = __logf(delta + 1.0f);
  float cs = 0.f;
#pragma unroll
  for (int z = 0; z < NZ_; ++z) cs += __cosf(gamma[z] * ld);
  M[idx] = decay * cs;
}

// ---------------- tiled bf16 WMMA GEMM: C[n,m] = sum_d A[n,d]*W[m,d] + bias[m]
// Block tile 256x64, 8 waves; each wave: 32 rows x 64 cols (8 accumulators),
// so each B fragment is reused by two A fragments (8 WMMA per 32-wide K step).
// mode 0: bf16 out, (b,h,s,dk) layout   (Q, K)
// mode 1: bf16 out, (b,h,dk,s) layout   (V transposed)
// mode 2: f32 out,  row-major (n, m)    (final projection)
__global__ __launch_bounds__(256)
void gemm_bf16(const __bf16* __restrict__ A, const __bf16* __restrict__ W,
               const float* __restrict__ bias, void* __restrict__ dst, int mode) {
  __shared__ __align__(32) __bf16 lA[256 * 32];
  __shared__ __align__(32) __bf16 lW[64 * 32];
  const int tid = threadIdx.x;
  const int wave = tid >> 5, lane = tid & 31, ln = lane & 15, lh = lane >> 4;
  const int rowbase = blockIdx.x * 256;
  const int colbase = blockIdx.y * 64;

  v8f acc[2][4];
#pragma unroll
  for (int u = 0; u < 2; ++u)
#pragma unroll
    for (int t = 0; t < 4; ++t)
#pragma unroll
      for (int e = 0; e < 8; ++e) acc[u][t][e] = 0.f;

  const int wrow = tid >> 2, wq4 = tid & 3;  // W tile: 64 rows x 4 quarters

  for (int k0 = 0; k0 < D_; k0 += 32) {
    // A tile 256x32: one full row (64B = 4x16B) per thread, async to LDS
    const __bf16* arow = &A[(size_t)(rowbase + tid) * D_ + k0];
#pragma unroll
    for (int c = 0; c < 4; ++c)
      async_copy16(arow + c * 8, &lA[tid * 32 + c * 8]);
    // W tile 64x32: 16B per thread
    async_copy16(&W[(size_t)(colbase + wrow) * D_ + k0 + wq4 * 8],
                 &lW[wrow * 32 + wq4 * 8]);
    if (k0 + 32 < D_)
      __builtin_prefetch(&A[(size_t)(rowbase + tid) * D_ + k0 + 32], 0, 1);
    async_wait_all();
    __syncthreads();

    // Two A fragments: rows wave*32 + u*16 + ln, 16 contiguous K at lh*16
    v16bf afrag[2];
#pragma unroll
    for (int u = 0; u < 2; ++u) {
      const int frow = wave * 32 + u * 16 + ln;
      afrag[u] = *(const v16bf*)&lA[frow * 32 + lh * 16];
    }
#pragma unroll
    for (int t = 0; t < 4; ++t) {
      // B fragment: N = ln, 16 contiguous K at lh*16 (reused by both A frags)
      v16bf bfrag = *(const v16bf*)&lW[(t * 16 + ln) * 32 + lh * 16];
#pragma unroll
      for (int u = 0; u < 2; ++u)
        acc[u][t] = __builtin_amdgcn_wmma_f32_16x16x32_bf16(
            false, afrag[u], false, bfrag, (short)0, acc[u][t], false, false);
    }
    __syncthreads();
  }

#pragma unroll
  for (int u = 0; u < 2; ++u)
#pragma unroll
    for (int t = 0; t < 4; ++t)
#pragma unroll
      for (int g = 0; g < 8; ++g) {
        int n = rowbase + wave * 32 + u * 16 + g + 8 * lh;  // C row = g + 8*lh
        int m = colbase + t * 16 + ln;                      // C col = ln
        float v = acc[u][t][g] + bias[m];
        int b = n >> 11, s = n & (S_ - 1);
        int h = m >> 6, dk = m & 63;
        if (mode == 0)
          ((__bf16*)dst)[((size_t)(b * H_ + h) * S_ + s) * DK_ + dk] = (__bf16)v;
        else if (mode == 1)
          ((__bf16*)dst)[((size_t)(b * H_ + h) * DK_ + dk) * S_ + s] = (__bf16)v;
        else
          ((float*)dst)[(size_t)n * D_ + m] = v;
      }
}

// ---------------- flash attention with M-kernel / omega / causal masking ----
// grid (S/128, H, B), 8 waves; each wave owns 16 query rows; j loop step 32.
__global__ __launch_bounds__(256)
void flash_attn(const __bf16* __restrict__ Q, const __bf16* __restrict__ K,
                const __bf16* __restrict__ VT, const float* __restrict__ Mker,
                const int* __restrict__ ids, const float* __restrict__ omega,
                __bf16* __restrict__ Obf) {
  __shared__ __align__(32) __bf16 lP[8 * 16 * 32];  // per-wave 16x32 P tile
  const int tid = threadIdx.x;
  const int wave = tid >> 5, lane = tid & 31, ln = lane & 15, lh = lane >> 4;
  const int b = blockIdx.z, h = blockIdx.y;
  const int rowbase = blockIdx.x * 128 + wave * 16;
  const size_t qkbase = (size_t)(b * H_ + h) * S_;   // rows of length DK
  const size_t vtbase = (size_t)(b * H_ + h) * DK_;  // rows of length S

  // Q fragments (A layout), rows = rowbase + ln, two 32-wide dk chunks
  v16bf qf[2];
#pragma unroll
  for (int c = 0; c < 2; ++c) {
    const __bf16* qr = &Q[(qkbase + rowbase + ln) * DK_ + c * 32];
    qf[c] = *(const v16bf*)&qr[lh * 16];
  }

  int tid_i[8];
  float m_run[8], l_run[8];
  v8f O[4];
#pragma unroll
  for (int g = 0; g < 8; ++g) {
    tid_i[g] = ids[b * S_ + rowbase + g + 8 * lh];
    m_run[g] = -3.0e38f;
    l_run[g] = 0.f;
  }
#pragma unroll
  for (int t = 0; t < 4; ++t)
#pragma unroll
    for (int e = 0; e < 8; ++e) O[t][e] = 0.f;

  const int jend = rowbase + 16;  // causal bound for this wave's rows
  for (int j0 = 0; j0 < jend; j0 += 32) {
    // ---- scores: 2 j-subtiles x 2 dk-chunks = 4 WMMAs
    v8f sc[2];
#pragma unroll
    for (int t = 0; t < 2; ++t) {
#pragma unroll
      for (int e = 0; e < 8; ++e) sc[t][e] = 0.f;
      const int j = j0 + t * 16 + ln;
#pragma unroll
      for (int c = 0; c < 2; ++c) {
        v16bf kf = *(const v16bf*)&K[(qkbase + j) * DK_ + c * 32 + lh * 16];
        sc[t] = __builtin_amdgcn_wmma_f32_16x16x32_bf16(
            false, qf[c], false, kf, (short)0, sc[t], false, false);
      }
    }
    // ---- scale, memory kernel, omega + causal masks
    float p[2][8], tmax[8];
#pragma unroll
    for (int g = 0; g < 8; ++g) tmax[g] = -3.0e38f;
#pragma unroll
    for (int t = 0; t < 2; ++t) {
      const int j = j0 + t * 16 + ln;
      const int tidj = ids[b * S_ + j];
#pragma unroll
      for (int g = 0; g < 8; ++g) {
        const int i = rowbase + g + 8 * lh;
        float v = sc[t][g] * 0.125f;  // 1/sqrt(DK)
        if (j > i) {
          v = -1.0e9f;
        } else {
          v *= Mker[(size_t)i * S_ + j];
          if (omega[(size_t)tid_i[g] * V_ + tidj] == 0.f) v = -1.0e9f;
        }
        p[t][g] = v;
        tmax[g] = fmaxf(tmax[g], v);
      }
    }
    // ---- online softmax (reductions across the 16 lanes of each half-wave)
    float scale[8];
#pragma unroll
    for (int g = 0; g < 8; ++g) {
#pragma unroll
      for (int off = 8; off >= 1; off >>= 1)
        tmax[g] = fmaxf(tmax[g], __shfl_xor(tmax[g], off, 32));
      float nm = fmaxf(m_run[g], tmax[g]);
      scale[g] = __expf(m_run[g] - nm);
      m_run[g] = nm;
    }
    float rs[8];
#pragma unroll
    for (int g = 0; g < 8; ++g) rs[g] = 0.f;
#pragma unroll
    for (int t = 0; t < 2; ++t)
#pragma unroll
      for (int g = 0; g < 8; ++g) {
        float e = __expf(p[t][g] - m_run[g]);
        p[t][g] = e;
        rs[g] += e;
      }
#pragma unroll
    for (int g = 0; g < 8; ++g) {
#pragma unroll
      for (int off = 8; off >= 1; off >>= 1)
        rs[g] += __shfl_xor(rs[g], off, 32);
      l_run[g] = l_run[g] * scale[g] + rs[g];
    }
    // ---- P: C-layout regs -> per-wave LDS tile -> A-layout fragment
    __bf16* pw = &lP[wave * 512];
#pragma unroll
    for (int t = 0; t < 2; ++t)
#pragma unroll
      for (int g = 0; g < 8; ++g)
        pw[(g + 8 * lh) * 32 + t * 16 + ln] = (__bf16)p[t][g];
#pragma unroll
    for (int t = 0; t < 4; ++t)
#pragma unroll
      for (int g = 0; g < 8; ++g) O[t][g] *= scale[g];
    v16bf pf = *(const v16bf*)&pw[ln * 32 + lh * 16];
    // ---- P @ V : 4 WMMAs, V fragments contiguous thanks to transposed layout
#pragma unroll
    for (int t = 0; t < 4; ++t) {
      v16bf vf = *(const v16bf*)&VT[(vtbase + t * 16 + ln) * S_ + j0 + lh * 16];
      O[t] = __builtin_amdgcn_wmma_f32_16x16x32_bf16(
          false, pf, false, vf, (short)0, O[t], false, false);
    }
  }
  // ---- normalize and store bf16 (b,s,d)
  float inv[8];
#pragma unroll
  for (int g = 0; g < 8; ++g) inv[g] = (l_run[g] > 0.f) ? 1.f / l_run[g] : 0.f;
#pragma unroll
  for (int t = 0; t < 4; ++t)
#pragma unroll
    for (int g = 0; g < 8; ++g) {
      const int i = rowbase + g + 8 * lh;
      const int dk = t * 16 + ln;
      Obf[(size_t)(b * S_ + i) * D_ + h * DK_ + dk] = (__bf16)(O[t][g] * inv[g]);
    }
}

extern "C" void kernel_launch(void* const* d_in, const int* in_sizes, int n_in,
                              void* d_out, int out_size, void* d_ws, size_t ws_size,
                              hipStream_t stream) {
  const float* x     = (const float*)d_in[0];
  const float* wq    = (const float*)d_in[1];
  const float* bq    = (const float*)d_in[2];
  const float* wk    = (const float*)d_in[3];
  const float* bk    = (const float*)d_in[4];
  const float* wv    = (const float*)d_in[5];
  const float* bv    = (const float*)d_in[6];
  const float* wo    = (const float*)d_in[7];
  const float* bo    = (const float*)d_in[8];
  const float* omega = (const float*)d_in[9];
  const float* gamma = (const float*)d_in[10];
  const int*   ids   = (const int*)d_in[11];
  // d_in[12] = causal_mask: causality handled analytically, not loaded.

  char* ws = (char*)d_ws;
  const size_t MB = 1024u * 1024u;
  __bf16* xb  = (__bf16*)(ws + 0);
  __bf16* wqb = (__bf16*)(ws + 8 * MB);
  __bf16* wkb = (__bf16*)(ws + 10 * MB);
  __bf16* wvb = (__bf16*)(ws + 12 * MB);
  __bf16* wob = (__bf16*)(ws + 14 * MB);
  __bf16* qb  = (__bf16*)(ws + 16 * MB);
  __bf16* kb  = (__bf16*)(ws + 24 * MB);
  __bf16* vtb = (__bf16*)(ws + 32 * MB);
  __bf16* ab  = (__bf16*)(ws + 40 * MB);
  float*  Mk  = (float*)(ws + 48 * MB);

  const int NX = B_ * S_ * D_;  // 4194304
  const int NW = D_ * D_;       // 1048576
  cvt_f32_bf16<<<(NX / 4 + 255) / 256, 256, 0, stream>>>(x, xb, NX / 4);
  cvt_f32_bf16<<<(NW / 4 + 255) / 256, 256, 0, stream>>>(wq, wqb, NW / 4);
  cvt_f32_bf16<<<(NW / 4 + 255) / 256, 256, 0, stream>>>(wk, wkb, NW / 4);
  cvt_f32_bf16<<<(NW / 4 + 255) / 256, 256, 0, stream>>>(wv, wvb, NW / 4);
  cvt_f32_bf16<<<(NW / 4 + 255) / 256, 256, 0, stream>>>(wo, wob, NW / 4);
  build_memkernel<<<(S_ * S_ + 255) / 256, 256, 0, stream>>>(gamma, Mk);

  dim3 gg(16, 16);  // 4096/256 x 1024/64
  gemm_bf16<<<gg, 256, 0, stream>>>(xb, wqb, bq, qb, 0);
  gemm_bf16<<<gg, 256, 0, stream>>>(xb, wkb, bk, kb, 0);
  gemm_bf16<<<gg, 256, 0, stream>>>(xb, wvb, bv, vtb, 1);

  dim3 ga(S_ / 128, H_, B_);
  flash_attn<<<ga, 256, 0, stream>>>(qb, kb, vtb, Mk, ids, omega, ab);

  gemm_bf16<<<gg, 256, 0, stream>>>(ab, wob, bo, d_out, 2);
}
